// GraphCrossAttn_37924561223998
// MI455X (gfx1250) — matrix operand order repaired and checked by reading, not compiled, not run on hardware
//
#include <hip/hip_runtime.h>

typedef __bf16 bf16_t;
typedef __attribute__((ext_vector_type(2)))  float  v2f;
typedef __attribute__((ext_vector_type(8)))  float  v8f;
typedef __attribute__((ext_vector_type(4)))  int    v4i;
typedef __attribute__((ext_vector_type(8)))  bf16_t v8bf;
typedef __attribute__((ext_vector_type(16))) bf16_t v16bf;

#define AS1 __attribute__((address_space(1)))
#define AS3 __attribute__((address_space(3)))
#define CDIV(a, b) (((a) + (b) - 1) / (b))

#if __has_builtin(__builtin_amdgcn_global_load_async_to_lds_b128) && \
    __has_builtin(__builtin_amdgcn_s_wait_asynccnt)
#define HAVE_ASYNC_LDS 1
#endif

// 16-byte global -> LDS copy (async path: GLOBAL_LOAD_ASYNC_TO_LDS_B128, ASYNCcnt-tracked)
__device__ __forceinline__ void copy16(const bf16_t* g, bf16_t* l) {
#ifdef HAVE_ASYNC_LDS
  __builtin_amdgcn_global_load_async_to_lds_b128((AS1 v4i*)g, (AS3 v4i*)l, 0, 0);
#else
  *(v8bf*)l = *(const v8bf*)g;
#endif
}
template <int N>
__device__ __forceinline__ void wait_async() {
#ifdef HAVE_ASYNC_LDS
  __builtin_amdgcn_s_wait_asynccnt((short)N);
#endif
}

// ---------------- ordered-uint encoding for float atomicMax ----------------
__device__ __forceinline__ unsigned fenc(float f) {
  unsigned u = __float_as_uint(f);
  return (u & 0x80000000u) ? ~u : (u | 0x80000000u);
}
__device__ __forceinline__ float fdec(unsigned u) {
  return (u & 0x80000000u) ? __uint_as_float(u & 0x7FFFFFFFu) : __uint_as_float(~u);
}

// ---------------- weight prep: transpose (+pad) ----------------
__global__ void k_wt_f32(const float* __restrict__ W, float* __restrict__ Wt, int K, int Nc) {
  int idx = blockIdx.x * blockDim.x + threadIdx.x;
  if (idx >= K * Nc) return;
  int n = idx / K, k = idx % K;
  Wt[idx] = W[(size_t)k * Nc + n];
}

__global__ void k_wt_bf16(const float* __restrict__ W, bf16_t* __restrict__ Wt,
                          int K, int Nc, int Kpad, int Npad) {
  int idx = blockIdx.x * blockDim.x + threadIdx.x;
  if (idx >= Kpad * Npad) return;
  int n = idx / Kpad, k = idx % Kpad;
  float v = (n < Nc && k < K) ? W[(size_t)k * Nc + n] : 0.f;
  Wt[idx] = (bf16_t)v;
}

// vec[k,h] = sum_c W[k, h*128+c] * att[h,c]
__global__ void k_collapse(const float* __restrict__ W, const float* __restrict__ att,
                           float* __restrict__ vec, int K) {
  int idx = blockIdx.x * blockDim.x + threadIdx.x;
  if (idx >= K * 4) return;
  int k = idx >> 2, h = idx & 3;
  const float* wr = W + (size_t)k * 512 + h * 128;
  const float* ar = att + h * 128;
  float s = 0.f;
  for (int c = 0; c < 128; ++c) s += wr[c] * ar[c];
  vec[idx] = s;
}

__global__ void k_scores(const float* __restrict__ act, int D,
                         const float* __restrict__ vs, const float* __restrict__ vd,
                         float* __restrict__ as_, float* __restrict__ ad_, int N) {
  int idx = blockIdx.x * blockDim.x + threadIdx.x;
  if (idx >= N * 4) return;
  int n = idx >> 2, h = idx & 3;
  const float* a = act + (size_t)n * D;
  float ss = 0.f, sd = 0.f;
  for (int k = 0; k < D; ++k) { float v = a[k]; ss += v * vs[k * 4 + h]; sd += v * vd[k * 4 + h]; }
  as_[idx] = ss; ad_[idx] = sd;
}

__global__ void k_init_nm(unsigned* __restrict__ menc, float* __restrict__ denom, int n) {
  int i = blockIdx.x * blockDim.x + threadIdx.x;
  if (i < n) { menc[i] = 0u; denom[i] = 0.f; }
}

// ---------------- GAT edge passes ----------------
__global__ void k_edge_score(const long long* __restrict__ src, const long long* __restrict__ dst,
                             const float* __restrict__ as_, const float* __restrict__ ad_,
                             float* __restrict__ alpha, unsigned* __restrict__ menc, int E) {
  int i = blockIdx.x * blockDim.x + threadIdx.x;
  if (i >= E * 4) return;
  int e = i >> 2, h = i & 3;
  int s = (int)src[e], d = (int)dst[e];
  float v = as_[s * 4 + h] + ad_[d * 4 + h];
  v = v > 0.f ? v : 0.2f * v;                 // leaky_relu(0.2)
  alpha[i] = v;
  atomicMax(menc + d * 4 + h, fenc(v));       // segment max
}

__global__ void k_edge_soft(const long long* __restrict__ dst, float* __restrict__ alpha,
                            const unsigned* __restrict__ menc, float* __restrict__ denom, int E) {
  int i = blockIdx.x * blockDim.x + threadIdx.x;
  if (i >= E * 4) return;
  int e = i >> 2, h = i & 3;
  int d = (int)dst[e];
  float w = __expf(alpha[i] - fdec(menc[d * 4 + h]));
  alpha[i] = w;
  atomicAdd(denom + d * 4 + h, w);            // segment sum
}

__global__ void k_edge_msg(const long long* __restrict__ src, const long long* __restrict__ dst,
                           const float* __restrict__ alpha, const float* __restrict__ denom,
                           const float* __restrict__ xs, float* __restrict__ hacc, int E) {
  int e = blockIdx.x;
  if (e >= E) return;
  int s = (int)src[e], d = (int)dst[e];
  int t = threadIdx.x;
#pragma unroll
  for (int rep = 0; rep < 2; ++rep) {
    int j = t + rep * 256;
    int h = j >> 7;
    float w = alpha[(size_t)e * 4 + h] / (denom[(size_t)d * 4 + h] + 1e-16f);
    atomicAdd(hacc + (size_t)d * 512 + j, xs[(size_t)s * 512 + j] * w);
  }
}

// ---------------- LayerNorm + ReLU (block per node, LDS reduce) ----------------
__global__ void k_ln_relu(const float* __restrict__ hin, const float* __restrict__ cb,
                          const float* __restrict__ g, const float* __restrict__ b,
                          float* __restrict__ hf, bf16_t* __restrict__ hb) {
  int n = blockIdx.x, t = threadIdx.x;
  const float* row = hin + (size_t)n * 512;
  float v[4], s = 0.f, s2 = 0.f;
#pragma unroll
  for (int i = 0; i < 4; ++i) {
    int j = t + i * 128;
    float x = row[j] + cb[j];
    v[i] = x; s += x; s2 += x * x;
  }
  __shared__ float rs[128], rq[128];
  rs[t] = s; rq[t] = s2; __syncthreads();
  for (int off = 64; off > 0; off >>= 1) {
    if (t < off) { rs[t] += rs[t + off]; rq[t] += rq[t + off]; }
    __syncthreads();
  }
  float mu  = rs[0] * (1.f / 512.f);
  float var = rq[0] * (1.f / 512.f) - mu * mu;
  float inv = rsqrtf(var + 1e-5f);
#pragma unroll
  for (int i = 0; i < 4; ++i) {
    int j = t + i * 128;
    float y = (v[i] - mu) * inv * g[j] + b[j];
    y = fmaxf(y, 0.f);
    hf[(size_t)n * 512 + j] = y;
    hb[(size_t)n * 512 + j] = (bf16_t)y;
  }
}

__global__ void k_addb_bf16(const float* __restrict__ h, const float* __restrict__ b,
                            bf16_t* __restrict__ o, int D, int total) {
  int i = blockIdx.x * blockDim.x + threadIdx.x;
  if (i >= total) return;
  o[i] = (bf16_t)(h[i] + b[i % D]);
}

// ---------------- bf16 WMMA GEMM: one wave = 16x64 tile, 4 accumulators ----------------
// A tile (16 x 32 per K-step) is staged into LDS via async-to-LDS (double buffered),
// shared across the 4 N-subtiles; B fragments stream from L2-resident transposed weights.
__global__ __launch_bounds__(32)
void k_gemm_bf16(const bf16_t* __restrict__ A, int lda,
                 const bf16_t* __restrict__ Wt, int ldw,
                 const float* __restrict__ bias,
                 float* __restrict__ Cf, bf16_t* __restrict__ Cb,
                 int ldc, int Nc, int K, int relu) {
  __shared__ __align__(16) bf16_t abuf[2][16 * 32];
  int lane = threadIdx.x & 31;
  int half = lane >> 4, lm = lane & 15;
  int m0 = blockIdx.y << 4;
  int nb0 = blockIdx.x << 6;                      // 64 output columns per wave
  const bf16_t* arow = A + (size_t)m0 * lda;
  const bf16_t* br[4];
#pragma unroll
  for (int j = 0; j < 4; ++j) br[j] = Wt + (size_t)(nb0 + j * 16 + lm) * ldw;

  v8f acc[4];
#pragma unroll
  for (int j = 0; j < 4; ++j) acc[j] = (v8f){0.f, 0.f, 0.f, 0.f, 0.f, 0.f, 0.f, 0.f};

  // stage one 16x32 A chunk (1 KB) into abuf[buf]: 64 16B chunks, 2 per lane
  auto stage = [&](int kbase, int buf) {
#pragma unroll
    for (int c = lane; c < 64; c += 32) {
      int r = c >> 2, q = (c & 3) * 8;
      copy16(arow + (size_t)r * lda + kbase + q, &abuf[buf][r * 32 + q]);
    }
  };

  stage(0, 0);
  for (int kk = 0; kk < K; kk += 32) {
    int cur = (kk >> 5) & 1;
    if (kk + 32 < K) {
      __builtin_prefetch(br[0] + kk + 64, 0, 1);   // global_prefetch_b8
      __builtin_prefetch(br[2] + kk + 64, 0, 1);
      stage(kk + 32, cur ^ 1);
      wait_async<2>();                             // 2 newest may be outstanding
    } else {
      wait_async<0>();
    }
    // A fragment from LDS (ds_load_b128 x2), contiguous K = 16*half + s (matches B)
    const bf16_t* lr = &abuf[cur][lm * 32];
    v8bf alo = *(const v8bf*)(lr + 16 * half);
    v8bf ahi = *(const v8bf*)(lr + 16 * half + 8);
    v16bf a = __builtin_shufflevector(alo, ahi, 0,1,2,3,4,5,6,7,8,9,10,11,12,13,14,15);
#pragma unroll
    for (int j = 0; j < 4; ++j) {                  // 4 independent WMMA chains
      v16bf bm = *(const v16bf*)(br[j] + kk + 16 * half);
      acc[j] = __builtin_amdgcn_wmma_f32_16x16x32_bf16(false, a, false, bm, (short)0,
                                                       acc[j], false, false);
    }
  }
#pragma unroll
  for (int j = 0; j < 4; ++j) {
    int n = nb0 + j * 16 + lm;
    if (n < Nc) {
      float bs = bias ? bias[n] : 0.f;
#pragma unroll
      for (int r = 0; r < 8; ++r) {
        int m = m0 + r + 8 * half;                 // C/D: VGPR r -> M=r (+8 hi lanes)
        float v = acc[j][r] + bs;
        if (relu) v = fmaxf(v, 0.f);
        if (Cf) Cf[(size_t)m * ldc + n] = v;
        if (Cb) Cb[(size_t)m * ldc + n] = (bf16_t)v;
      }
    }
  }
}

// ---------------- fp32 WMMA 16x16x4 for input-embedding GEMMs (16x64 per wave) --------
__global__ __launch_bounds__(32)
void k_gemm_f32(const float* __restrict__ A, int lda, int acol,
                const float* __restrict__ Wt, int ldw,
                const float* __restrict__ bias,
                float* __restrict__ Cf, bf16_t* __restrict__ Cb,
                int ldc, int ccol, int K) {
  int lane = threadIdx.x & 31;
  int half = lane >> 4, lm = lane & 15;
  int m0 = blockIdx.y << 4;
  int nb0 = blockIdx.x << 6;
  const float* arow = A + (size_t)(m0 + lm) * lda + acol;
  const float* br[4];
#pragma unroll
  for (int j = 0; j < 4; ++j) br[j] = Wt + (size_t)(nb0 + j * 16 + lm) * ldw;

  v8f acc[4];
#pragma unroll
  for (int j = 0; j < 4; ++j) acc[j] = (v8f){0.f, 0.f, 0.f, 0.f, 0.f, 0.f, 0.f, 0.f};

  for (int k = 0; k < K; k += 4) {
    __builtin_prefetch(arow + k + 64, 0, 1);
    v2f a = *(const v2f*)(arow + k + 2 * half);    // A 16x4: lane K = 2h, 2h+1
#pragma unroll
    for (int j = 0; j < 4; ++j) {
      v2f b = *(const v2f*)(br[j] + k + 2 * half); // B 4x16 from transposed W
      acc[j] = __builtin_amdgcn_wmma_f32_16x16x4_f32(false, a, false, b, (short)0,
                                                     acc[j], false, false);
    }
  }
#pragma unroll
  for (int j = 0; j < 4; ++j) {
    int n = nb0 + j * 16 + lm;
    float bs = bias[n];
#pragma unroll
    for (int r = 0; r < 8; ++r) {
      int m = m0 + r + 8 * half;
      float v = acc[j][r] + bs;
      Cf[(size_t)m * ldc + ccol + n] = v;
      Cb[(size_t)m * ldc + ccol + n] = (bf16_t)v;
    }
  }
}

// ---------------- host side ----------------
extern "C" void kernel_launch(void* const* d_in, const int* in_sizes, int n_in,
                              void* d_out, int out_size, void* d_ws, size_t ws_size,
                              hipStream_t stream) {
  constexpr int RNA = 2000, PROT = 100;
  const float*     x  = (const float*)d_in[0];
  const long long* ei = (const long long*)d_in[1];
  const float* W_rna = (const float*)d_in[2];  const float* b_rna = (const float*)d_in[3];
  const float* W_prot= (const float*)d_in[4];  const float* b_prot= (const float*)d_in[5];
  const float* c1_Ws = (const float*)d_in[6];  const float* c1_Wd = (const float*)d_in[7];
  const float* c1_as = (const float*)d_in[8];  const float* c1_ad = (const float*)d_in[9];
  const float* c1_b  = (const float*)d_in[10];
  const float* l1_W  = (const float*)d_in[11]; const float* l1_b  = (const float*)d_in[12];
  const float* ln_g  = (const float*)d_in[13]; const float* ln_b  = (const float*)d_in[14];
  const float* c2_Ws = (const float*)d_in[15]; const float* c2_Wd = (const float*)d_in[16];
  const float* c2_as = (const float*)d_in[17]; const float* c2_ad = (const float*)d_in[18];
  const float* c2_b  = (const float*)d_in[19];
  const float* l2_W  = (const float*)d_in[20]; const float* l2_b  = (const float*)d_in[21];
  const float* agg_W = (const float*)d_in[22]; const float* agg_b = (const float*)d_in[23];
  const float* dr_W  = (const float*)d_in[24]; const float* dr_b  = (const float*)d_in[25];
  const float* dp_W  = (const float*)d_in[26]; const float* dp_b  = (const float*)d_in[27];
  const float* rr_W  = (const float*)d_in[28]; const float* rr_b  = (const float*)d_in[29];
  const float* rp_W  = (const float*)d_in[30]; const float* rp_b  = (const float*)d_in[31];

  const int Nn = in_sizes[0] / (RNA + PROT);   // 10000 (multiple of 16)
  const int E  = in_sizes[1] / 2;              // 160000
  const long long* src = ei;
  const long long* dst = ei + E;

  float* out_rna  = (float*)d_out;                       // [N, 2000]
  float* out_prot = out_rna + (size_t)Nn * RNA;          // [N, 100]
  float* out_emb  = out_prot + (size_t)Nn * PROT;        // [N, 128]

  // ---- carve workspace ----
  char* p = (char*)d_ws;
  auto carve = [&](size_t bytes) -> char* {
    char* r = p; p += (bytes + 255) & ~(size_t)255; return r;
  };
  float*   wt_rna  = (float*)  carve(sizeof(float) * 128 * 2000);
  float*   wt_prot = (float*)  carve(sizeof(float) * 128 * 100);
  bf16_t*  wt_c1ws = (bf16_t*) carve(2ull * 512 * 256);
  bf16_t*  wt_l1   = (bf16_t*) carve(2ull * 512 * 256);
  bf16_t*  wt_c2ws = (bf16_t*) carve(2ull * 512 * 512);
  bf16_t*  wt_l2   = (bf16_t*) carve(2ull * 512 * 512);
  bf16_t*  wt_agg  = (bf16_t*) carve(2ull * 128 * 512);
  bf16_t*  wt_dr   = (bf16_t*) carve(2ull * 128 * 128);
  bf16_t*  wt_dp   = (bf16_t*) carve(2ull * 128 * 128);
  bf16_t*  wt_rr   = (bf16_t*) carve(2ull * 2048 * 128);   // Npad 2048 (64-col waves)
  bf16_t*  wt_rp   = (bf16_t*) carve(2ull * 128 * 128);    // Npad 128
  float*   vec_s1  = (float*)  carve(sizeof(float) * 256 * 4);
  float*   vec_d1  = (float*)  carve(sizeof(float) * 256 * 4);
  float*   vec_s2  = (float*)  carve(sizeof(float) * 512 * 4);
  float*   vec_d2  = (float*)  carve(sizeof(float) * 512 * 4);
  float*   xin     = (float*)  carve(4ull * Nn * 256);
  bf16_t*  xin_bf  = (bf16_t*) carve(2ull * Nn * 256);
  float*   a_s     = (float*)  carve(4ull * Nn * 4);
  float*   a_d     = (float*)  carve(4ull * Nn * 4);
  unsigned* menc   = (unsigned*)carve(4ull * Nn * 4);
  float*   denom   = (float*)  carve(4ull * Nn * 4);
  float*   alpha   = (float*)  carve(4ull * E * 4);
  float*   xs      = (float*)  carve(4ull * Nn * 512);
  float*   hacc    = (float*)  carve(4ull * Nn * 512);
  float*   h1      = (float*)  carve(4ull * Nn * 512);
  bf16_t*  h1_bf   = (bf16_t*) carve(2ull * Nn * 512);
  bf16_t*  h2_bf   = (bf16_t*) carve(2ull * Nn * 512);
  bf16_t*  emb_bf  = (bf16_t*) carve(2ull * Nn * 128);
  bf16_t*  tr_bf   = (bf16_t*) carve(2ull * Nn * 128);
  bf16_t*  tp_bf   = (bf16_t*) carve(2ull * Nn * 128);

  const int mb = Nn / 16;   // 625 M-tiles

  // ---- weight prep ----
  k_wt_f32 <<<CDIV(128 * 2000, 256), 256, 0, stream>>>(W_rna,  wt_rna,  2000, 128);
  k_wt_f32 <<<CDIV(128 * 100, 256),  256, 0, stream>>>(W_prot, wt_prot, 100,  128);
  k_wt_bf16<<<CDIV(512 * 256, 256),  256, 0, stream>>>(c1_Ws, wt_c1ws, 256, 512, 256, 512);
  k_wt_bf16<<<CDIV(512 * 256, 256),  256, 0, stream>>>(l1_W,  wt_l1,   256, 512, 256, 512);
  k_wt_bf16<<<CDIV(512 * 512, 256),  256, 0, stream>>>(c2_Ws, wt_c2ws, 512, 512, 512, 512);
  k_wt_bf16<<<CDIV(512 * 512, 256),  256, 0, stream>>>(l2_W,  wt_l2,   512, 512, 512, 512);
  k_wt_bf16<<<CDIV(128 * 512, 256),  256, 0, stream>>>(agg_W, wt_agg,  512, 128, 512, 128);
  k_wt_bf16<<<CDIV(128 * 128, 256),  256, 0, stream>>>(dr_W,  wt_dr,   128, 128, 128, 128);
  k_wt_bf16<<<CDIV(128 * 128, 256),  256, 0, stream>>>(dp_W,  wt_dp,   128, 128, 128, 128);
  k_wt_bf16<<<CDIV(2048 * 128, 256), 256, 0, stream>>>(rr_W,  wt_rr,   128, 2000, 128, 2048);
  k_wt_bf16<<<CDIV(128 * 128, 256),  256, 0, stream>>>(rp_W,  wt_rp,   128, 100,  128, 128);
  k_collapse<<<CDIV(256 * 4, 256), 256, 0, stream>>>(c1_Ws, c1_as, vec_s1, 256);
  k_collapse<<<CDIV(256 * 4, 256), 256, 0, stream>>>(c1_Wd, c1_ad, vec_d1, 256);
  k_collapse<<<CDIV(512 * 4, 256), 256, 0, stream>>>(c2_Ws, c2_as, vec_s2, 512);
  k_collapse<<<CDIV(512 * 4, 256), 256, 0, stream>>>(c2_Wd, c2_ad, vec_d2, 512);

  // ---- input embedding: xin = [x@W_rna+b | x@W_prot+b]  (fp32 WMMA) ----
  k_gemm_f32<<<dim3(2, mb), 32, 0, stream>>>(x, RNA + PROT, 0,   wt_rna,  2000, b_rna,
                                             xin, xin_bf, 256, 0,   2000);
  k_gemm_f32<<<dim3(2, mb), 32, 0, stream>>>(x, RNA + PROT, RNA, wt_prot, 100,  b_prot,
                                             xin, xin_bf, 256, 128, 100);

  // ---- GAT block 1 ----
  k_scores<<<CDIV(Nn * 4, 256), 256, 0, stream>>>(xin, 256, vec_s1, vec_d1, a_s, a_d, Nn);
  k_gemm_bf16<<<dim3(8, mb), 32, 0, stream>>>(xin_bf, 256, wt_c1ws, 256, nullptr,
                                              xs, nullptr, 512, 512, 256, 0);
  k_gemm_bf16<<<dim3(8, mb), 32, 0, stream>>>(xin_bf, 256, wt_l1, 256, l1_b,
                                              hacc, nullptr, 512, 512, 256, 0);
  k_init_nm  <<<CDIV(Nn * 4, 256), 256, 0, stream>>>(menc, denom, Nn * 4);
  k_edge_score<<<CDIV(E * 4, 256), 256, 0, stream>>>(src, dst, a_s, a_d, alpha, menc, E);
  k_edge_soft <<<CDIV(E * 4, 256), 256, 0, stream>>>(dst, alpha, menc, denom, E);
  k_edge_msg  <<<dim3(E), 256, 0, stream>>>(src, dst, alpha, denom, xs, hacc, E);
  k_ln_relu   <<<dim3(Nn), 128, 0, stream>>>(hacc, c1_b, ln_g, ln_b, h1, h1_bf);

  // ---- GAT block 2 ----
  k_scores<<<CDIV(Nn * 4, 256), 256, 0, stream>>>(h1, 512, vec_s2, vec_d2, a_s, a_d, Nn);
  k_gemm_bf16<<<dim3(8, mb), 32, 0, stream>>>(h1_bf, 512, wt_c2ws, 512, nullptr,
                                              xs, nullptr, 512, 512, 512, 0);
  k_gemm_bf16<<<dim3(8, mb), 32, 0, stream>>>(h1_bf, 512, wt_l2, 512, l2_b,
                                              hacc, nullptr, 512, 512, 512, 0);
  k_init_nm  <<<CDIV(Nn * 4, 256), 256, 0, stream>>>(menc, denom, Nn * 4);
  k_edge_score<<<CDIV(E * 4, 256), 256, 0, stream>>>(src, dst, a_s, a_d, alpha, menc, E);
  k_edge_soft <<<CDIV(E * 4, 256), 256, 0, stream>>>(dst, alpha, menc, denom, E);
  k_edge_msg  <<<dim3(E), 256, 0, stream>>>(src, dst, alpha, denom, xs, hacc, E);
  k_addb_bf16 <<<CDIV(Nn * 512, 256), 256, 0, stream>>>(hacc, c2_b, h2_bf, 512, Nn * 512);

  // ---- heads: emb + reconstructions ----
  k_gemm_bf16<<<dim3(2, mb), 32, 0, stream>>>(h2_bf, 512, wt_agg, 512, agg_b,
                                              out_emb, emb_bf, 128, 128, 512, 1 /*relu*/);
  k_gemm_bf16<<<dim3(2, mb), 32, 0, stream>>>(emb_bf, 128, wt_dr, 128, dr_b,
                                              nullptr, tr_bf, 128, 128, 128, 0);
  k_gemm_bf16<<<dim3(32, mb), 32, 0, stream>>>(tr_bf, 128, wt_rr, 128, rr_b,
                                               out_rna, nullptr, 2000, 2000, 128, 0);
  k_gemm_bf16<<<dim3(2, mb), 32, 0, stream>>>(emb_bf, 128, wt_dp, 128, dp_b,
                                              nullptr, tp_bf, 128, 128, 128, 0);
  k_gemm_bf16<<<dim3(2, mb), 32, 0, stream>>>(tp_bf, 128, wt_rp, 128, rp_b,
                                              out_prot, nullptr, 100, 100, 128, 0);
}
